// TextImplicitFactorizationModule_50929722196540
// MI455X (gfx1250) — matrix operand order repaired and checked by reading, not compiled, not run on hardware
//
#include <hip/hip_runtime.h>
#include <hip/hip_bf16.h>
#include <hip/hip_fp16.h>

typedef _Float16 h16;
typedef __attribute__((ext_vector_type(16))) _Float16 v16h;
typedef __attribute__((ext_vector_type(8)))  _Float16 v8h;
typedef __attribute__((ext_vector_type(8)))  float    v8f;

#define B_SZ   64
#define S_SZ   1024
#define EMB_SZ 128
#define HID_SZ 256
#define NF_SZ  8
#define VF_SZ  128
#define EF_SZ  32
#define JD     512   // 2*HID (joined width)

// ---------------------------------------------------------------------------
// WMMA helpers (gfx1250, wave32). D = A(16x32 f16) * B(32x16 f16) + C(16x16 f32)
// ---------------------------------------------------------------------------
__device__ __forceinline__ v8f wmma_f16(v16h a, v16h b, v8f c) {
  return __builtin_amdgcn_wmma_f32_16x16x32_f16(false, a, false, b, (short)0, c,
                                                false, false);
}

// A fragment: 16x32 f16 tile from row-major [16 x ld] buffer at k-offset k0.
// A layout (matches B): lanes 0-15: K=0..15, lanes 16-31: K=16..31
__device__ __forceinline__ v16h load_a_frag(const h16* base, int ld, int k0, int lane) {
  int row = lane & 15;
  int kb  = (lane & 16) ? 16 : 0;
  const h16* p = base + row * ld + k0 + kb;
  return *(const v16h*)(p);
}

// B fragment: B[k][n] = W[n0+n][k0+k] with W row-major (stride ldw halfs).
// Lanes 0-15: K=0..15 of column N=lane, lanes 16-31: K=16..31 of column N=lane-16.
__device__ __forceinline__ v16h load_b_frag(const h16* w_n0, int ldw, int k0, int lane) {
  int n  = lane & 15;
  int kb = (lane & 16) ? 16 : 0;
  return *(const v16h*)(w_n0 + (size_t)n * ldw + k0 + kb);
}

__device__ __forceinline__ float sigm(float x) { return 1.f / (1.f + __expf(-x)); }

// ---------------------------------------------------------------------------
// f32 -> f16 weight conversion
// ---------------------------------------------------------------------------
__global__ void f32_to_f16_kernel(const float* __restrict__ src,
                                  h16* __restrict__ dst, int n) {
  int i = blockIdx.x * 256 + threadIdx.x;
  if (i < n) dst[i] = (h16)src[i];
}

// ---------------------------------------------------------------------------
// Bi-LSTM: grid = 8 blocks (dir in {0,1} x 4 batch-row tiles of 16).
// Block = 512 threads = 16 waves; wave w owns the single 16-wide h-column tile
// [16w, 16w+16) and all four gates for it -> cell update is wave-local, no
// private arrays, ~90 VGPRs live (no spills, no vgpr-msb).
// Per step per wave: (4 + 8) k-steps x 4 gates = 48 WMMA; K loops kept rolled
// (#pragma unroll 1) so the allocator sees a bounded load window; latency is
// hidden by 4 waves/SIMD, not by in-wave unrolling.
// ---------------------------------------------------------------------------
__global__ __launch_bounds__(512) void lstm_kernel(
    const int* __restrict__ tokens, const float* __restrict__ emb_table,
    const h16* __restrict__ Wih_f_h, const h16* __restrict__ Whh_f_h,
    const float* __restrict__ b_f,
    const h16* __restrict__ Wih_b_h, const h16* __restrict__ Whh_b_h,
    const float* __restrict__ b_b,
    h16* __restrict__ joined_h) {
  __shared__ alignas(32) h16 x_h[16 * EMB_SZ];   // 4 KB, current x tile (f16)
  __shared__ alignas(32) h16 h_h[16 * HID_SZ];   // 8 KB, current h state (f16)

  const int dir = blockIdx.x >> 2;           // 0 = forward, 1 = backward
  const int m0  = (blockIdx.x & 3) * 16;     // batch row base
  const h16*  Wih  = dir ? Wih_b_h : Wih_f_h;
  const h16*  Whh  = dir ? Whh_b_h : Whh_f_h;
  const float* bias = dir ? b_b : b_f;

  const int tid  = threadIdx.x;
  const int wave = tid >> 5;         // 0..15
  const int lane = tid & 31;
  const int hc0  = wave * 16;        // this wave's h-column tile base
  const int ncol = lane & 15;
  const int col  = hc0 + ncol;       // this lane's h column
  const int mrow = (lane & 16) ? 8 : 0;

  // Per-gate weight row bases (constant across time)
  const h16* WihI = Wih + (size_t)(0 * HID_SZ + hc0) * EMB_SZ;
  const h16* WihF = Wih + (size_t)(1 * HID_SZ + hc0) * EMB_SZ;
  const h16* WihG = Wih + (size_t)(2 * HID_SZ + hc0) * EMB_SZ;
  const h16* WihO = Wih + (size_t)(3 * HID_SZ + hc0) * EMB_SZ;
  const h16* WhhI = Whh + (size_t)(0 * HID_SZ + hc0) * HID_SZ;
  const h16* WhhF = Whh + (size_t)(1 * HID_SZ + hc0) * HID_SZ;
  const h16* WhhG = Whh + (size_t)(2 * HID_SZ + hc0) * HID_SZ;
  const h16* WhhO = Whh + (size_t)(3 * HID_SZ + hc0) * HID_SZ;

  const float bi0 = bias[0 * HID_SZ + col];
  const float bf0 = bias[1 * HID_SZ + col];
  const float bg0 = bias[2 * HID_SZ + col];
  const float bo0 = bias[3 * HID_SZ + col];

  for (int i = tid; i < 16 * HID_SZ; i += 512) h_h[i] = (h16)0.f;

  float cst[8];
#pragma unroll
  for (int r = 0; r < 8; ++r) cst[r] = 0.f;

  for (int t = 0; t < S_SZ; ++t) {
    const int tt = dir ? (S_SZ - 1 - t) : t;

    // Stage x_t (16 rows x 128) into LDS: wave w stages row w (4 f32 per lane).
    {
      int row = wave;
      int c0  = lane * 4;
      int tok = tokens[(m0 + row) * S_SZ + tt];
      const float* e = emb_table + (size_t)tok * EMB_SZ + c0;
#pragma unroll
      for (int q = 0; q < 4; ++q) x_h[row * EMB_SZ + c0 + q] = (h16)e[q];
    }
    __syncthreads();

    v8f ai, af, ag, ao;
#pragma unroll
    for (int r = 0; r < 8; ++r) { ai[r] = bi0; af[r] = bf0; ag[r] = bg0; ao[r] = bo0; }

    // x @ Wih^T  (K = 128)
#pragma unroll 1
    for (int kk = 0; kk < EMB_SZ; kk += 32) {
      v16h a = load_a_frag(x_h, EMB_SZ, kk, lane);
      ai = wmma_f16(a, load_b_frag(WihI, EMB_SZ, kk, lane), ai);
      af = wmma_f16(a, load_b_frag(WihF, EMB_SZ, kk, lane), af);
      ag = wmma_f16(a, load_b_frag(WihG, EMB_SZ, kk, lane), ag);
      ao = wmma_f16(a, load_b_frag(WihO, EMB_SZ, kk, lane), ao);
    }
    // h @ Whh^T  (K = 256)
#pragma unroll 1
    for (int kk = 0; kk < HID_SZ; kk += 32) {
      v16h a = load_a_frag(h_h, HID_SZ, kk, lane);
      ai = wmma_f16(a, load_b_frag(WhhI, HID_SZ, kk, lane), ai);
      af = wmma_f16(a, load_b_frag(WhhF, HID_SZ, kk, lane), af);
      ag = wmma_f16(a, load_b_frag(WhhG, HID_SZ, kk, lane), ag);
      ao = wmma_f16(a, load_b_frag(WhhO, HID_SZ, kk, lane), ao);
    }

    // LSTM cell update (wave-local; C-fragment layout)
    float hv[8];
#pragma unroll
    for (int r = 0; r < 8; ++r) {
      float cn = sigm(af[r]) * cst[r] + sigm(ai[r]) * tanhf(ag[r]);
      cst[r] = cn;
      hv[r]  = sigm(ao[r]) * tanhf(cn);
    }
    __syncthreads();   // all waves finished reading h_h / x_h

    // Write new h to LDS (next step's A operand) and to the global joined buf.
#pragma unroll
    for (int r = 0; r < 8; ++r) {
      int M = mrow + r;
      h16 hh = (h16)hv[r];
      h_h[M * HID_SZ + col] = hh;
      joined_h[((size_t)(m0 + M) * S_SZ + tt) * JD + dir * HID_SZ + col] = hh;
    }
    __syncthreads();
  }
}

// ---------------------------------------------------------------------------
// Fused attention scores: scores[b,f,s] =
//   key2_b[f] + sum_h key2_W[f,h] * tanh( sum_d joined[b,s,d]*key1_W[f,h,d] + key1_b[f,h] )
// GEMM M=65536 (b*s), N=2048 (f*h), K=512 — the 137 GFLOP hot spot, all WMMA.
// grid = (2048 row-tiles of 32, 8 factors), block = 256 (8 waves).
// A tile (32 x 512) staged once into LDS (shared by all 8 waves); each wave
// owns 32 h-columns (2 N-tiles) x 2 M-tiles = 4 accumulators, B fragments
// reused across both M-tiles. K loop rolled to bound register pressure.
// ---------------------------------------------------------------------------
__global__ __launch_bounds__(256) void attn_scores_kernel(
    const h16* __restrict__ joined_h, const h16* __restrict__ key1_h,
    const float* __restrict__ key1_b, const float* __restrict__ key2_W,
    const float* __restrict__ key2_b, float* __restrict__ scores) {
  const int mt   = blockIdx.x;              // 32-row tile of M = b*S + s
  const int f    = blockIdx.y;
  const int tid  = threadIdx.x;
  const int wave = tid >> 5;
  const int lane = tid & 31;

  __shared__ alignas(32) h16 a_tile[32 * JD];   // 32 KB
  __shared__ float spart[32][8];

  // Cooperative stage of the 32x512 A tile (16B per thread per iteration).
  {
    const h16* src = joined_h + (size_t)mt * 32 * JD;
#pragma unroll
    for (int it = 0; it < 8; ++it) {
      int i = it * (256 * 8) + tid * 8;
      *(v8h*)(a_tile + i) = *(const v8h*)(src + i);
    }
  }
  __syncthreads();

  const int n0 = wave * 32;                     // this wave's h-column base
  const h16* B0 = key1_h + ((size_t)f * HID_SZ + n0) * JD;
  const h16* B1 = B0 + (size_t)16 * JD;

  v8f acc00, acc01, acc10, acc11;               // [mtile][jt]
#pragma unroll
  for (int r = 0; r < 8; ++r) {
    acc00[r] = 0.f; acc01[r] = 0.f; acc10[r] = 0.f; acc11[r] = 0.f;
  }

#pragma unroll 1
  for (int kk = 0; kk < JD; kk += 32) {
    v16h a0 = load_a_frag(a_tile,            JD, kk, lane);
    v16h a1 = load_a_frag(a_tile + 16 * JD,  JD, kk, lane);
    v16h b0 = load_b_frag(B0, JD, kk, lane);
    v16h b1 = load_b_frag(B1, JD, kk, lane);
    acc00 = wmma_f16(a0, b0, acc00);
    acc10 = wmma_f16(a1, b0, acc10);
    acc01 = wmma_f16(a0, b1, acc01);
    acc11 = wmma_f16(a1, b1, acc11);
  }

  // Epilogue: hid = tanh(acc + key1_b), val = hid * key2_W, reduce over N.
  float part0[8], part1[8];                     // per M-tile row partials
#pragma unroll
  for (int r = 0; r < 8; ++r) { part0[r] = 0.f; part1[r] = 0.f; }

  {
    const float kb0 = key1_b[f * HID_SZ + n0 + (lane & 15)];
    const float kw0 = key2_W[f * HID_SZ + n0 + (lane & 15)];
    const float kb1 = key1_b[f * HID_SZ + n0 + 16 + (lane & 15)];
    const float kw1 = key2_W[f * HID_SZ + n0 + 16 + (lane & 15)];
#pragma unroll
    for (int r = 0; r < 8; ++r) {
      float v0 = tanhf(acc00[r] + kb0) * kw0 + tanhf(acc01[r] + kb1) * kw1;
      float v1 = tanhf(acc10[r] + kb0) * kw0 + tanhf(acc11[r] + kb1) * kw1;
      v0 += __shfl_xor(v0, 1, 32);
      v0 += __shfl_xor(v0, 2, 32);
      v0 += __shfl_xor(v0, 4, 32);
      v0 += __shfl_xor(v0, 8, 32);
      v1 += __shfl_xor(v1, 1, 32);
      v1 += __shfl_xor(v1, 2, 32);
      v1 += __shfl_xor(v1, 4, 32);
      v1 += __shfl_xor(v1, 8, 32);
      part0[r] += v0;
      part1[r] += v1;
    }
  }
  // lane 0 holds rows 0..7, lane 16 holds rows 8..15 (per M-tile)
  if ((lane & 15) == 0) {
    int mbase = (lane & 16) ? 8 : 0;
#pragma unroll
    for (int r = 0; r < 8; ++r) {
      spart[mbase + r][wave]      = part0[r];
      spart[16 + mbase + r][wave] = part1[r];
    }
  }
  __syncthreads();
  if (tid < 32) {
    float s = key2_b[f];
#pragma unroll
    for (int w = 0; w < 8; ++w) s += spart[tid][w];
    int M  = mt * 32 + tid;
    int b  = M >> 10;          // / S
    int si = M & (S_SZ - 1);
    scores[((size_t)b * NF_SZ + f) * S_SZ + si] = s;
  }
}

// ---------------------------------------------------------------------------
// softmax over s + mean_state[b,f,d] = sum_s attn[s]*joined[b,s,d]
// grid = 512 (b,f), block = 256. Memory-bound; ~0.5 GB total reads.
// ---------------------------------------------------------------------------
__global__ __launch_bounds__(256) void attn_apply_kernel(
    const float* __restrict__ scores, const h16* __restrict__ joined_h,
    float* __restrict__ mean_state) {
  const int bf = blockIdx.x;
  const int b  = bf >> 3;
  const int tid = threadIdx.x;
  __shared__ float attn[S_SZ];
  __shared__ float red[256];

  float mx = -1e30f;
  for (int s = tid; s < S_SZ; s += 256) {
    float v = scores[(size_t)bf * S_SZ + s];
    attn[s] = v;
    mx = fmaxf(mx, v);
  }
  red[tid] = mx; __syncthreads();
  for (int st = 128; st > 0; st >>= 1) {
    if (tid < st) red[tid] = fmaxf(red[tid], red[tid + st]);
    __syncthreads();
  }
  mx = red[0]; __syncthreads();

  float sum = 0.f;
  for (int s = tid; s < S_SZ; s += 256) {
    float e = __expf(attn[s] - mx);
    attn[s] = e;
    sum += e;
  }
  red[tid] = sum; __syncthreads();
  for (int st = 128; st > 0; st >>= 1) {
    if (tid < st) red[tid] += red[tid + st];
    __syncthreads();
  }
  float inv = 1.f / red[0]; __syncthreads();
  for (int s = tid; s < S_SZ; s += 256) attn[s] *= inv;
  __syncthreads();

  float m0 = 0.f, m1 = 0.f;
  const h16* jb = joined_h + (size_t)b * S_SZ * JD;
  for (int s = 0; s < S_SZ; ++s) {
    float a = attn[s];
    m0 += a * (float)jb[(size_t)s * JD + tid];
    m1 += a * (float)jb[(size_t)s * JD + tid + 256];
  }
  mean_state[(size_t)bf * JD + tid]       = m0;
  mean_state[(size_t)bf * JD + tid + 256] = m1;
}

// ---------------------------------------------------------------------------
// Factor head: logits (dot-512) -> softmax(128) -> mean_factor (dot-128)
// grid = 512 (b,f), block = 128. Tiny (~70 MFLOP total).
// ---------------------------------------------------------------------------
__global__ __launch_bounds__(128) void head_kernel(
    const float* __restrict__ mean_state, const float* __restrict__ flw_W,
    const float* __restrict__ flw_b, const float* __restrict__ fv_W,
    const float* __restrict__ fv_b, float* __restrict__ out) {
  const int bf = blockIdx.x;
  const int b  = bf >> 3;
  const int f  = bf & 7;
  const int tid = threadIdx.x;
  __shared__ float ms[JD];
  __shared__ float probs[VF_SZ];
  __shared__ float red[128];

  for (int d = tid; d < JD; d += 128) ms[d] = mean_state[(size_t)bf * JD + d];
  __syncthreads();

  const float* wv = flw_W + ((size_t)f * VF_SZ + tid) * JD;
  float lg = flw_b[f * VF_SZ + tid];
  for (int d = 0; d < JD; ++d) lg += ms[d] * wv[d];

  red[tid] = lg; __syncthreads();
  for (int st = 64; st > 0; st >>= 1) {
    if (tid < st) red[tid] = fmaxf(red[tid], red[tid + st]);
    __syncthreads();
  }
  float mx = red[0]; __syncthreads();
  float e = __expf(lg - mx);
  red[tid] = e; __syncthreads();
  for (int st = 64; st > 0; st >>= 1) {
    if (tid < st) red[tid] += red[tid + st];
    __syncthreads();
  }
  float inv = 1.f / red[0]; __syncthreads();
  probs[tid] = e * inv;
  __syncthreads();

  if (tid < EF_SZ) {
    float acc = fv_b[f * EF_SZ + tid];
    const float* w = fv_W + ((size_t)f * EF_SZ + tid) * VF_SZ;
    for (int v = 0; v < VF_SZ; ++v) acc += probs[v] * w[v];
    out[(size_t)b * (NF_SZ * EF_SZ) + f * EF_SZ + tid] = acc;
  }
}

// ---------------------------------------------------------------------------
// Host launch
// ---------------------------------------------------------------------------
extern "C" void kernel_launch(void* const* d_in, const int* in_sizes, int n_in,
                              void* d_out, int out_size, void* d_ws, size_t ws_size,
                              hipStream_t stream) {
  const int*   tokens    = (const int*)  d_in[0];
  const float* emb_table = (const float*)d_in[1];
  const float* Wih_f     = (const float*)d_in[2];
  const float* Whh_f     = (const float*)d_in[3];
  const float* b_f       = (const float*)d_in[4];
  const float* Wih_b     = (const float*)d_in[5];
  const float* Whh_b     = (const float*)d_in[6];
  const float* b_b       = (const float*)d_in[7];
  const float* key1_W    = (const float*)d_in[8];
  const float* key1_b    = (const float*)d_in[9];
  const float* key2_W    = (const float*)d_in[10];
  const float* key2_b    = (const float*)d_in[11];
  const float* flw_W     = (const float*)d_in[12];
  const float* flw_b     = (const float*)d_in[13];
  const float* fv_W      = (const float*)d_in[14];
  const float* fv_b      = (const float*)d_in[15];
  float* out = (float*)d_out;

  // Workspace carve-out (256B aligned)
  char*  ws  = (char*)d_ws;
  size_t off = 0;
  auto carve = [&](size_t bytes) -> void* {
    void* p = ws + off;
    off = (off + bytes + 255) & ~(size_t)255;
    return p;
  };
  const size_t nWih  = 4 * HID_SZ * EMB_SZ;            // 131072
  const size_t nWhh  = 4 * HID_SZ * HID_SZ;            // 262144
  const size_t nKey1 = (size_t)NF_SZ * HID_SZ * JD;    // 1048576
  h16* Wih_f_h = (h16*)carve(nWih  * 2);
  h16* Whh_f_h = (h16*)carve(nWhh  * 2);
  h16* Wih_b_h = (h16*)carve(nWih  * 2);
  h16* Whh_b_h = (h16*)carve(nWhh  * 2);
  h16* key1_h  = (h16*)carve(nKey1 * 2);
  h16* joined  = (h16*)carve((size_t)B_SZ * S_SZ * JD * 2);           // 64 MB
  float* scores     = (float*)carve((size_t)B_SZ * NF_SZ * S_SZ * 4); // 2 MB
  float* mean_state = (float*)carve((size_t)B_SZ * NF_SZ * JD * 4);   // 1 MB

  // 1) weight conversion to f16
  auto cvt = [&](const float* s, h16* d, size_t n) {
    f32_to_f16_kernel<<<dim3((unsigned)((n + 255) / 256)), 256, 0, stream>>>(s, d, (int)n);
  };
  cvt(Wih_f, Wih_f_h, nWih);
  cvt(Whh_f, Whh_f_h, nWhh);
  cvt(Wih_b, Wih_b_h, nWih);
  cvt(Whh_b, Whh_b_h, nWhh);
  cvt(key1_W, key1_h, nKey1);

  // 2) bidirectional LSTM (8 independent recurrences: 2 dir x 4 batch tiles)
  lstm_kernel<<<8, 512, 0, stream>>>(tokens, emb_table,
                                     Wih_f_h, Whh_f_h, b_f,
                                     Wih_b_h, Whh_b_h, b_b, joined);

  // 3) fused attention-score GEMM (137 GFLOP, WMMA)
  attn_scores_kernel<<<dim3(B_SZ * S_SZ / 32, NF_SZ), 256, 0, stream>>>(
      joined, key1_h, key1_b, key2_W, key2_b, scores);

  // 4) softmax + attention-weighted mean state
  attn_apply_kernel<<<B_SZ * NF_SZ, 256, 0, stream>>>(scores, joined, mean_state);

  // 5) factor heads -> output embeddings (B, NF*EF)
  head_kernel<<<B_SZ * NF_SZ, 128, 0, stream>>>(mean_state, flw_W, flw_b,
                                                fv_W, fv_b, out);
}
